// SelfAttentionHead_137438953881
// MI455X (gfx1250) — matrix operand order repaired and checked by reading, not compiled, not run on hardware
//
#include <hip/hip_runtime.h>

typedef __bf16 bf16_t;
typedef __attribute__((ext_vector_type(16))) __bf16 v16bf;
typedef __attribute__((ext_vector_type(8)))  __bf16 v8bf;
typedef __attribute__((ext_vector_type(8)))  float  v8f;
typedef __attribute__((ext_vector_type(4)))  unsigned int u32x4;
typedef __attribute__((ext_vector_type(8)))  int i32x8;
typedef __attribute__((ext_vector_type(4)))  int i32x4;

#define B_N  8
#define T_N  2048
#define D_N  1024
#define HD_N 64
#define NEG_INF (-1.0e30f)

// ---------------------------------------------------------------------------
// TDM: async-load a [32 rows x 64 cols] bf16 tile (row-major, row stride 64)
// from global into LDS. D# per CDNA5 ISA.
// Group0: count=1 | lds_addr | global_addr[56:0] | type=2.
// Group1: data_size=2B, tensor_dim0=64, tensor_dim1=2048, tile=64x32,
//         tensor_dim0_stride=64. Remaining groups zero (2D tensor).
// Toolchain here is the 6-arg clang-23 form:
//   (u32x4 g0, i32x8 g1, i32x4, i32x4, i32x8, i32 cpol)
// ---------------------------------------------------------------------------
__device__ __forceinline__ void tdm_load_tile_32x64_bf16(const bf16_t* gptr,
                                                         bf16_t* lptr) {
    unsigned long long ga = (unsigned long long)(uintptr_t)gptr;
    unsigned lds = (unsigned)(uintptr_t)lptr;      // low 32 bits = LDS byte addr
    u32x4 g0;
    g0[0] = 1u;                                    // count=1, user desc
    g0[1] = lds;                                   // lds_addr
    g0[2] = (unsigned)ga;                          // global_addr[31:0]
    g0[3] = (unsigned)(ga >> 32) | (2u << 30);     // global_addr[56:32] | type=2
    i32x8 g1;
    g1[0] = 0x00010000;                            // data_size=1 (2 bytes)
    g1[1] = (int)(64u << 16);                      // tensor_dim0 = 64 (lo16)
    g1[2] = (int)(2048u << 16);                    // dim0 hi=0 | tensor_dim1 lo16
    g1[3] = (int)(64u << 16);                      // dim1 hi=0 | tile_dim0 = 64
    g1[4] = 32;                                    // tile_dim1 = 32
    g1[5] = 64;                                    // tensor_dim0_stride lo32
    g1[6] = 0;
    g1[7] = 0;                                     // dim1_stride unused (2D)
    i32x4 gz4 = {0, 0, 0, 0};
    i32x8 gz8 = {0, 0, 0, 0, 0, 0, 0, 0};
    __builtin_amdgcn_tensor_load_to_lds(g0, g1, gz4, gz4, gz8, 0);
}

// ---------------------------------------------------------------------------
// Kernel 1: qkv = x @ [Wq|Wk|Wv], f32 in -> bf16 out; V stored transposed.
// One wave per 16-row tile of x (flat over B*T). 8 waves / block.
// ---------------------------------------------------------------------------
__global__ __launch_bounds__(256)
void qkv_proj_kernel(const float* __restrict__ x,
                     const float* __restrict__ Wq,
                     const float* __restrict__ Wk,
                     const float* __restrict__ Wv,
                     bf16_t* __restrict__ qws,
                     bf16_t* __restrict__ kws,
                     bf16_t* __restrict__ vtws)
{
    // W chunk staged transposed: ldsW[m][h*32 + d_local], bf16
    __shared__ bf16_t ldsW[3][HD_N * 32];

    const int tid  = threadIdx.x;
    const int wave = tid >> 5;
    const int lane = tid & 31;
    const int hi   = lane >> 4;   // half-wave id
    const int lm   = lane & 15;
    const int tile = blockIdx.x * 8 + wave;      // 0..1023
    const long row0 = (long)tile * 16;           // flat row over B*T

    v8f acc[3][4] = {};                          // 3 matrices x 4 N-tiles of 16

    const int khalf = hi * 8;
    const int kb    = hi * 16;

    for (int kc = 0; kc < D_N / 32; ++kc) {
        __syncthreads();
        // Cooperative stage of W[kc*32 .. +31][0..63] for all three matrices,
        // transposed + converted to bf16 (B-fragments become contiguous).
        #pragma unroll
        for (int j = 0; j < 8; ++j) {
            int idx = tid + j * 256;             // 0..2047
            int dl  = idx >> 6;                  // 0..31
            int h   = idx & 63;
            int g   = (kc * 32 + dl) * HD_N + h;
            ldsW[0][h * 32 + dl] = (bf16_t)Wq[g];
            ldsW[1][h * 32 + dl] = (bf16_t)Wk[g];
            ldsW[2][h * 32 + dl] = (bf16_t)Wv[g];
        }
        __syncthreads();

        // A fragment of x (16x32, bf16): row = lm, contiguous K, k = 16*hi + i.
        const float* xrow = x + (row0 + lm) * D_N + (long)kc * 32;
        // prefetch next x chunk into cache (global_prefetch_b8)
        if (kc + 1 < D_N / 32) __builtin_prefetch(xrow + 32, 0, 1);
        v16bf a;
        #pragma unroll
        for (int i = 0; i < 8; ++i) a[i]     = (bf16_t)xrow[kb + i];
        #pragma unroll
        for (int i = 0; i < 8; ++i) a[8 + i] = (bf16_t)xrow[kb + 8 + i];

        #pragma unroll
        for (int m = 0; m < 3; ++m) {
            // preload all 4 B fragments, then issue 4 WMMAs back-to-back
            v16bf bfr[4];
            #pragma unroll
            for (int nt = 0; nt < 4; ++nt)
                bfr[nt] = *(const v16bf*)&ldsW[m][(nt * 16 + lm) * 32 + kb];
            #pragma unroll
            for (int nt = 0; nt < 4; ++nt)
                acc[m][nt] = __builtin_amdgcn_wmma_f32_16x16x32_bf16(
                    false, a, false, bfr[nt], (short)0, acc[m][nt], false, false);
        }
    }

    // Store: C layout row = r + 8*hi, col = nt*16 + lm.
    const long bb    = row0 / T_N;
    const long trow0 = row0 % T_N;
    #pragma unroll
    for (int nt = 0; nt < 4; ++nt) {
        #pragma unroll
        for (int r = 0; r < 8; ++r) {
            long mrow = row0 + r + 8 * hi;
            int  h    = nt * 16 + lm;
            qws[mrow * HD_N + h] = (bf16_t)acc[0][nt][r];
            kws[mrow * HD_N + h] = (bf16_t)acc[1][nt][r];
            long trow = trow0 + r + 8 * hi;
            vtws[(bb * HD_N + h) * T_N + trow] = (bf16_t)acc[2][nt][r];  // V^T
        }
    }
}

// ---------------------------------------------------------------------------
// Kernel 2: causal flash attention. One wave per 16-query tile; K tiles are
// TDM double-buffered into LDS (tensor_load_to_lds + s_wait_tensorcnt);
// QK^T and PV on v_wmma_f32_16x16x32_bf16; online softmax in f32.
// ---------------------------------------------------------------------------
__global__ __launch_bounds__(256)
void flash_attn_kernel(const bf16_t* __restrict__ qws,
                       const bf16_t* __restrict__ kws,
                       const bf16_t* __restrict__ vtws,
                       float* __restrict__ out)
{
    __shared__ bf16_t ldsK[8][2][32 * HD_N];     // per-wave double-buffered K
    __shared__ bf16_t ldsP[8][16 * 32];          // per-wave P tile (bf16)

    const int tid  = threadIdx.x;
    const int wave = tid >> 5;
    const int lane = tid & 31;
    const int hi   = lane >> 4;
    const int lm   = lane & 15;
    const int tile = blockIdx.x * 8 + wave;      // 0..1023
    const int bb   = tile >> 7;                  // batch
    const int t0   = (tile & 127) * 16;          // query-tile start within T
    const long row0 = (long)tile * 16;           // flat q/out row

    const int khalf = hi * 8;
    const int kb    = hi * 16;

    // Q A-fragments for both 32-wide h chunks (HD=64), loaded once.
    v16bf aq[2];
    #pragma unroll
    for (int hc = 0; hc < 2; ++hc) {
        const bf16_t* qrow = qws + (row0 + lm) * HD_N + hc * 32;
        v8bf lo = *(const v8bf*)(qrow + kb);
        v8bf up = *(const v8bf*)(qrow + kb + 8);
        #pragma unroll
        for (int i = 0; i < 8; ++i) { aq[hc][i] = lo[i]; aq[hc][8 + i] = up[i]; }
    }

    v8f acc[4] = {};
    float mrow[8], lrow[8];
    #pragma unroll
    for (int r = 0; r < 8; ++r) { mrow[r] = NEG_INF; lrow[r] = 0.0f; }

    const float scale = 0.125f * 1.44269504088896340736f;  // hd^-0.5 * log2(e)
    bf16_t* pl = &ldsP[wave][0];
    const bf16_t* kbase = kws + (long)bb * T_N * HD_N;

    // Prime the TDM pipeline: chunk 0 -> buffer 0.
    int buf = 0;
    tdm_load_tile_32x64_bf16(kbase, &ldsK[wave][0][0]);

    for (int s0 = 0; s0 <= t0 + 15; s0 += 32) {
        // ---- issue next K-tile TDM; wait for current tile -----------------
        const bool more = (s0 + 32 <= t0 + 15);
        if (more) {
            tdm_load_tile_32x64_bf16(kbase + (long)(s0 + 32) * HD_N,
                                     &ldsK[wave][buf ^ 1][0]);
            __builtin_amdgcn_s_wait_tensorcnt(1);   // current tile complete
        } else {
            __builtin_amdgcn_s_wait_tensorcnt(0);   // drain
        }
        asm volatile("" ::: "memory");              // fence ds reads below
        const bf16_t* kl = &ldsK[wave][buf][0];

        // ---- S = Q K^T, 16x32 tile as two 16x16 C fragments ---------------
        v8f sfrag[2] = {};
        #pragma unroll
        for (int sh = 0; sh < 2; ++sh) {
            #pragma unroll
            for (int hc = 0; hc < 2; ++hc) {
                v16bf bk = *(const v16bf*)(kl + (sh * 16 + lm) * HD_N
                                              + hc * 32 + kb);
                sfrag[sh] = __builtin_amdgcn_wmma_f32_16x16x32_bf16(
                    false, aq[hc], false, bk, (short)0, sfrag[sh], false, false);
            }
        }

        // ---- scale + causal mask ------------------------------------------
        const bool needMask = (s0 + 31 > t0);
        float p[2][8];
        #pragma unroll
        for (int sh = 0; sh < 2; ++sh) {
            #pragma unroll
            for (int r = 0; r < 8; ++r) {
                float v = sfrag[sh][r] * scale;
                if (needMask) {
                    int scol = s0 + sh * 16 + lm;
                    int mr   = t0 + r + 8 * hi;
                    if (scol > mr) v = NEG_INF;
                }
                p[sh][r] = v;
            }
        }

        // ---- online softmax (row stats across 16-lane halves) -------------
        #pragma unroll
        for (int r = 0; r < 8; ++r) {
            float tm = fmaxf(p[0][r], p[1][r]);
            tm = fmaxf(tm, __shfl_xor(tm, 1));
            tm = fmaxf(tm, __shfl_xor(tm, 2));
            tm = fmaxf(tm, __shfl_xor(tm, 4));
            tm = fmaxf(tm, __shfl_xor(tm, 8));
            float nm   = fmaxf(mrow[r], tm);
            float corr = exp2f(mrow[r] - nm);
            mrow[r] = nm;
            float e0 = exp2f(p[0][r] - nm);
            float e1 = exp2f(p[1][r] - nm);
            p[0][r] = e0; p[1][r] = e1;
            float rs = e0 + e1;
            rs += __shfl_xor(rs, 1);
            rs += __shfl_xor(rs, 2);
            rs += __shfl_xor(rs, 4);
            rs += __shfl_xor(rs, 8);
            lrow[r] = lrow[r] * corr + rs;
            #pragma unroll
            for (int nt = 0; nt < 4; ++nt) acc[nt][r] *= corr;
        }

        // ---- C-layout -> A-layout via per-wave LDS bounce -----------------
        #pragma unroll
        for (int sh = 0; sh < 2; ++sh)
            #pragma unroll
            for (int r = 0; r < 8; ++r)
                pl[(r + 8 * hi) * 32 + sh * 16 + lm] = (bf16_t)p[sh][r];

        v16bf ap;
        {
            v8bf lo = *(const v8bf*)(pl + lm * 32 + kb);
            v8bf up = *(const v8bf*)(pl + lm * 32 + kb + 8);
            #pragma unroll
            for (int i = 0; i < 8; ++i) { ap[i] = lo[i]; ap[8 + i] = up[i]; }
        }

        // ---- O += P V  (V^T layout makes B-fragments contiguous) ----------
        #pragma unroll
        for (int nt = 0; nt < 4; ++nt) {
            const bf16_t* vp = vtws + ((long)(bb * HD_N + nt * 16 + lm)) * T_N
                                      + s0 + kb;
            v16bf bv = *(const v16bf*)vp;
            acc[nt] = __builtin_amdgcn_wmma_f32_16x16x32_bf16(
                false, ap, false, bv, (short)0, acc[nt], false, false);
        }

        buf ^= 1;
    }

    // ---- epilogue: normalize and store f32 --------------------------------
    #pragma unroll
    for (int nt = 0; nt < 4; ++nt)
        #pragma unroll
        for (int r = 0; r < 8; ++r)
            out[(row0 + r + 8 * hi) * HD_N + nt * 16 + lm] =
                acc[nt][r] / lrow[r];
}

// ---------------------------------------------------------------------------
extern "C" void kernel_launch(void* const* d_in, const int* in_sizes, int n_in,
                              void* d_out, int out_size, void* d_ws, size_t ws_size,
                              hipStream_t stream) {
    const float* x  = (const float*)d_in[0];
    const float* Wq = (const float*)d_in[1];
    const float* Wk = (const float*)d_in[2];
    const float* Wv = (const float*)d_in[3];
    float* out = (float*)d_out;

    const long QKV_ELEMS = (long)B_N * T_N * HD_N;       // 1,048,576
    bf16_t* qws  = (bf16_t*)d_ws;
    bf16_t* kws  = qws + QKV_ELEMS;
    bf16_t* vtws = kws + QKV_ELEMS;                      // total 6 MB of ws

    // 1024 wave-tiles of 16 rows, 8 waves (256 threads) per block.
    qkv_proj_kernel<<<128, 256, 0, stream>>>(x, Wq, Wk, Wv, qws, kws, vtws);
    flash_attn_kernel<<<128, 256, 0, stream>>>(qws, kws, vtws, out);
}
